// Block_80015240724876
// MI455X (gfx1250) — compile-verified, matrix operand order checked by reading, not run on hardware
//
#include <hip/hip_runtime.h>

typedef unsigned short u16;
typedef __attribute__((ext_vector_type(16))) __bf16 v16bf;
typedef __attribute__((ext_vector_type(8)))  float  v8f;

constexpr int Bb  = 2;
constexpr int Tt  = 1024;
constexpr int Dd  = 768;
constexpr int Hh  = 12;
constexpr int HSz = 64;
constexpr int Ee  = 8;
constexpr int DFf = 3072;
constexpr int Nn  = Bb * Tt;      // 2048 tokens
constexpr int NCc = 3 * Dd;       // 2304 (q|k|v columns)

__device__ __forceinline__ u16 f2bf(float f) {
    unsigned int u = __float_as_uint(f);
    unsigned int r = (u + 0x7FFFu + ((u >> 16) & 1u)) >> 16;
    return (u16)r;
}
__device__ __forceinline__ float bf2f(u16 b) {
    return __uint_as_float(((unsigned int)b) << 16);
}

struct Frag {
    union { v16bf v; unsigned int d[8]; u16 u[16]; };
};

// Load one 16x32 bf16 A/B fragment row: per-lane K mapping per CDNA5 ISA
// (VGPR0..3: K = half*8 + {0..7}; VGPR4..7: K = 16 + half*8 + {0..7}).
// `row` points at the start of this lane's matrix row (A: row m, B^T: row n).
// Lowers to 2x global_load_b128 per fragment (compiler-verified round 1).
__device__ __forceinline__ void load_frag(const u16* __restrict__ row, Frag& f) {
    const int hh = (threadIdx.x >> 4) & 1;
#pragma unroll
    for (int j = 0; j < 8; ++j) {
        const int k0 = (j < 4) ? (hh * 8 + 2 * j) : (16 + hh * 8 + 2 * (j - 4));
        f.d[j] = *(const unsigned int*)(row + k0);
    }
}

__device__ __forceinline__ v8f wmma_bf(const Frag& a, const Frag& b, v8f c) {
    return __builtin_amdgcn_wmma_f32_16x16x32_bf16(
        false, a.v, false, b.v, (short)0, c, false, false);
}

// ---------------- weight prep: f32 -> bf16, transposed ----------------

__global__ void wt_qkv_kernel(const float* __restrict__ Wq, const float* __restrict__ Wk,
                              const float* __restrict__ Wv, u16* __restrict__ wqkvT) {
    long long idx = (long long)blockIdx.x * blockDim.x + threadIdx.x;
    if (idx >= (long long)NCc * Dd) return;
    int n = (int)(idx / Dd), d = (int)(idx % Dd);
    int reg = n / 768;
    int within = n - reg * 768;
    int h = within / HSz, hs = within % HSz;
    const float* src = (reg == 0) ? Wq : ((reg == 1) ? Wk : Wv);
    wqkvT[(size_t)n * Dd + d] = f2bf(src[((size_t)h * Dd + d) * HSz + hs]);
}

__global__ void wt_transpose_kernel(const float* __restrict__ src, u16* __restrict__ dst,
                                    int R, int C) {
    int b = blockIdx.y;
    long long idx = (long long)blockIdx.x * blockDim.x + threadIdx.x;
    if (idx >= (long long)R * C) return;
    int r = (int)(idx / C), c = (int)(idx % C);
    dst[((size_t)b * C + c) * R + r] = f2bf(src[((size_t)b * R + r) * C + c]);
}

// ---------------- layernorm: f32 rows -> bf16 rows ----------------

__global__ __launch_bounds__(256)
void ln_kernel(const float* __restrict__ src, const float* __restrict__ g,
               const float* __restrict__ bsh, u16* __restrict__ dst) {
    const int row = blockIdx.x;
    const float* s = src + (size_t)row * Dd;
    __shared__ float red[256];
    float loc = 0.f;
    for (int i = threadIdx.x; i < Dd; i += 256) loc += s[i];
    red[threadIdx.x] = loc; __syncthreads();
    for (int st = 128; st > 0; st >>= 1) {
        if (threadIdx.x < st) red[threadIdx.x] += red[threadIdx.x + st];
        __syncthreads();
    }
    float mu = red[0] / Dd; __syncthreads();
    loc = 0.f;
    for (int i = threadIdx.x; i < Dd; i += 256) { float dd = s[i] - mu; loc += dd * dd; }
    red[threadIdx.x] = loc; __syncthreads();
    for (int st = 128; st > 0; st >>= 1) {
        if (threadIdx.x < st) red[threadIdx.x] += red[threadIdx.x + st];
        __syncthreads();
    }
    float rstd = rsqrtf(red[0] / Dd + 1e-5f);
    for (int i = threadIdx.x; i < Dd; i += 256)
        dst[(size_t)row * Dd + i] = f2bf((s[i] - mu) * rstd * g[i] + bsh[i]);
}

// ---------------- QKV GEMM: 64x64 tile/wave, [2048x768]x[768x2304], scatter q/k/vT -------

__global__ __launch_bounds__(32)
void gemm_qkv(const u16* __restrict__ h1, const u16* __restrict__ wqkvT,
              u16* __restrict__ q, u16* __restrict__ k, u16* __restrict__ vT) {
    const int lane = threadIdx.x & 31, nl = lane & 15, hh = lane >> 4;
    const int tn = blockIdx.x, tm = blockIdx.y;     // 36 x 32
    const u16* arow[4];
    const u16* brow[4];
#pragma unroll
    for (int i = 0; i < 4; ++i) {
        arow[i] = h1 + (size_t)(tm * 64 + i * 16 + nl) * Dd;
        brow[i] = wqkvT + (size_t)(tn * 64 + i * 16 + nl) * Dd;
    }
    v8f c[4][4] = {};
    for (int kt = 0; kt < Dd; kt += 32) {
        Frag a[4], bfr[4];
#pragma unroll
        for (int i = 0; i < 4; ++i) load_frag(arow[i] + kt, a[i]);
#pragma unroll
        for (int i = 0; i < 4; ++i) load_frag(brow[i] + kt, bfr[i]);
        __builtin_prefetch(brow[0] + kt + 32, 0, 1);
#pragma unroll
        for (int mi = 0; mi < 4; ++mi)
#pragma unroll
            for (int ni = 0; ni < 4; ++ni)
                c[mi][ni] = wmma_bf(a[mi], bfr[ni], c[mi][ni]);
    }
#pragma unroll
    for (int ni = 0; ni < 4; ++ni) {
        const int n = tn * 64 + ni * 16 + nl;
        const int reg = n / 768, within = n % 768;
        const int h = within / HSz, hs = within % HSz;
#pragma unroll
        for (int mi = 0; mi < 4; ++mi) {
#pragma unroll
            for (int r = 0; r < 8; ++r) {
                int token = tm * 64 + mi * 16 + 8 * hh + r;
                int b = token >> 10, t = token & 1023;
                u16 val = f2bf(c[mi][ni][r]);
                size_t bh = (size_t)b * Hh + h;
                if (reg == 0)      q[(bh * Tt + t) * HSz + hs] = val;
                else if (reg == 1) k[(bh * Tt + t) * HSz + hs] = val;
                else               vT[(bh * HSz + hs) * Tt + t] = val;
            }
        }
    }
}

// ---------------- flash attention: 16-query tile per wave, 32-key blocks ----------------

__global__ __launch_bounds__(32)
void attn_kernel(const u16* __restrict__ q, const u16* __restrict__ k,
                 const u16* __restrict__ vT, u16* __restrict__ attout) {
    const float SCALE = 0.03608439182435161f;   // 768^-0.5 (reference scales by D)
    const int lane = threadIdx.x & 31, nl = lane & 15, hh = lane >> 4;
    const int qtile = blockIdx.x;               // 0..63
    const int bh = blockIdx.y;                  // 0..23
    const int b = bh / Hh, h = bh % Hh;
    __shared__ u16 ldsP[16 * 32];

    const u16* qrow = q + ((size_t)bh * Tt + qtile * 16 + nl) * HSz;
    Frag aq0, aq1;
    load_frag(qrow, aq0);
    load_frag(qrow + 32, aq1);

    v8f o0 = {}, o1 = {}, o2 = {}, o3 = {};
    float mrow[8], lrow[8];
#pragma unroll
    for (int r = 0; r < 8; ++r) { mrow[r] = -1e30f; lrow[r] = 0.f; }

    const int kbend = (qtile * 16 + 15) >> 5;
    for (int kb = 0; kb <= kbend; ++kb) {
        Frag bk;
        v8f s0 = {}, s1 = {};
        const u16* krow0 = k + ((size_t)bh * Tt + kb * 32 + nl) * HSz;
        load_frag(krow0, bk);       s0 = wmma_bf(aq0, bk, s0);
        load_frag(krow0 + 32, bk);  s0 = wmma_bf(aq1, bk, s0);
        const u16* krow1 = krow0 + 16 * HSz;
        load_frag(krow1, bk);       s1 = wmma_bf(aq0, bk, s1);
        load_frag(krow1 + 32, bk);  s1 = wmma_bf(aq1, bk, s1);

#pragma unroll
        for (int r = 0; r < 8; ++r) {
            const int row = qtile * 16 + 8 * hh + r;
            const int c0 = kb * 32 + nl, c1 = c0 + 16;
            float x0 = s0[r] * SCALE; if (c0 > row) x0 = -1e30f;
            float x1 = s1[r] * SCALE; if (c1 > row) x1 = -1e30f;
            float vmax = fmaxf(x0, x1);
#pragma unroll
            for (int msk = 8; msk >= 1; msk >>= 1)
                vmax = fmaxf(vmax, __shfl_xor(vmax, msk, 32));
            float mnew  = fmaxf(mrow[r], vmax);
            float alpha = __expf(mrow[r] - mnew);
            float p0 = __expf(x0 - mnew), p1 = __expf(x1 - mnew);
            float ps = p0 + p1;
#pragma unroll
            for (int msk = 8; msk >= 1; msk >>= 1)
                ps += __shfl_xor(ps, msk, 32);
            lrow[r] = lrow[r] * alpha + ps;
            mrow[r] = mnew;
            o0[r] *= alpha; o1[r] *= alpha; o2[r] *= alpha; o3[r] *= alpha;
            ldsP[(8 * hh + r) * 32 + nl]      = f2bf(p0);
            ldsP[(8 * hh + r) * 32 + 16 + nl] = f2bf(p1);
        }
        __syncthreads();
        Frag ap;
        load_frag(&ldsP[nl * 32], ap);
        __syncthreads();

        Frag bv;
        const u16* vbase = vT + (size_t)bh * HSz * Tt + kb * 32;
        load_frag(vbase + (size_t)(0 * 16 + nl) * Tt, bv); o0 = wmma_bf(ap, bv, o0);
        load_frag(vbase + (size_t)(1 * 16 + nl) * Tt, bv); o1 = wmma_bf(ap, bv, o1);
        load_frag(vbase + (size_t)(2 * 16 + nl) * Tt, bv); o2 = wmma_bf(ap, bv, o2);
        load_frag(vbase + (size_t)(3 * 16 + nl) * Tt, bv); o3 = wmma_bf(ap, bv, o3);
    }

#pragma unroll
    for (int r = 0; r < 8; ++r) {
        float inv = 1.0f / lrow[r];
        int token = b * Tt + qtile * 16 + 8 * hh + r;
        u16* orow = attout + (size_t)token * Dd + h * HSz;
        orow[0 * 16 + nl] = f2bf(o0[r] * inv);
        orow[1 * 16 + nl] = f2bf(o1[r] * inv);
        orow[2 * 16 + nl] = f2bf(o2[r] * inv);
        orow[3 * 16 + nl] = f2bf(o3[r] * inv);
    }
}

// ---------------- projection GEMM + residual: 64x64 tile/wave, x1 = x + att@Wp + bp -------

__global__ __launch_bounds__(32)
void gemm_proj(const u16* __restrict__ att, const u16* __restrict__ wpT,
               const float* __restrict__ x, const float* __restrict__ bp,
               float* __restrict__ x1) {
    const int lane = threadIdx.x & 31, nl = lane & 15, hh = lane >> 4;
    const int tn = blockIdx.x, tm = blockIdx.y;     // 12 x 32
    const u16* arow[4];
    const u16* brow[4];
#pragma unroll
    for (int i = 0; i < 4; ++i) {
        arow[i] = att + (size_t)(tm * 64 + i * 16 + nl) * Dd;
        brow[i] = wpT + (size_t)(tn * 64 + i * 16 + nl) * Dd;
    }
    v8f c[4][4] = {};
    for (int kt = 0; kt < Dd; kt += 32) {
        Frag a[4], bfr[4];
#pragma unroll
        for (int i = 0; i < 4; ++i) load_frag(arow[i] + kt, a[i]);
#pragma unroll
        for (int i = 0; i < 4; ++i) load_frag(brow[i] + kt, bfr[i]);
        __builtin_prefetch(brow[0] + kt + 32, 0, 1);
#pragma unroll
        for (int mi = 0; mi < 4; ++mi)
#pragma unroll
            for (int ni = 0; ni < 4; ++ni)
                c[mi][ni] = wmma_bf(a[mi], bfr[ni], c[mi][ni]);
    }
#pragma unroll
    for (int ni = 0; ni < 4; ++ni) {
        const int n = tn * 64 + ni * 16 + nl;
        const float bias = bp[n];
#pragma unroll
        for (int mi = 0; mi < 4; ++mi) {
#pragma unroll
            for (int r = 0; r < 8; ++r) {
                int token = tm * 64 + mi * 16 + 8 * hh + r;
                x1[(size_t)token * Dd + n] =
                    c[mi][ni][r] + bias + x[(size_t)token * Dd + n];
            }
        }
    }
}

// ---------------- gate: top-1 expert per token, atomic-append lists ----------------

__global__ __launch_bounds__(256)
void gate_kernel(const u16* __restrict__ h2, const float* __restrict__ Wg,
                 int* __restrict__ counts, int* __restrict__ list) {
    int tok = blockIdx.x * blockDim.x + threadIdx.x;
    if (tok >= Nn) return;
    float acc[Ee];
#pragma unroll
    for (int e = 0; e < Ee; ++e) acc[e] = 0.f;
    const u16* hrow = h2 + (size_t)tok * Dd;
    for (int d = 0; d < Dd; ++d) {
        float hv = bf2f(hrow[d]);
        const float* wrow = Wg + (size_t)d * Ee;
#pragma unroll
        for (int e = 0; e < Ee; ++e) acc[e] += hv * wrow[e];
    }
    int best = 0; float bv = acc[0];
#pragma unroll
    for (int e = 1; e < Ee; ++e)
        if (acc[e] > bv) { bv = acc[e]; best = e; }
    int pos = atomicAdd(&counts[best], 1);
    list[best * Nn + pos] = tok;
}

// ---------------- MoE GEMM1: 16x64 tile/wave, hid = relu(h2[tok] @ W1[e] + b1[e]) --------

__global__ __launch_bounds__(32)
void moe_ffn1(const u16* __restrict__ h2, const u16* __restrict__ w1T,
              const float* __restrict__ b1, const int* __restrict__ counts,
              const int* __restrict__ list, u16* __restrict__ hid) {
    const int tn = blockIdx.x;                 // 0..47 (DF/64)
    const int e = blockIdx.y >> 7;
    const int slot = blockIdx.y & 127;
    const int cnt = counts[e];
    if (slot * 16 >= cnt) return;
    const int lane = threadIdx.x & 31, nl = lane & 15, hh = lane >> 4;
    const int ridx = slot * 16 + nl;
    const int tokA = list[e * Nn + ((ridx < cnt) ? ridx : slot * 16)];
    const u16* arow = h2 + (size_t)tokA * Dd;
    const u16* brow[4];
#pragma unroll
    for (int i = 0; i < 4; ++i)
        brow[i] = w1T + ((size_t)e * DFf + tn * 64 + i * 16 + nl) * Dd;
    v8f c[4] = {};
    for (int kt = 0; kt < Dd; kt += 32) {
        Frag a, bfr[4];
        load_frag(arow + kt, a);
#pragma unroll
        for (int i = 0; i < 4; ++i) load_frag(brow[i] + kt, bfr[i]);
        __builtin_prefetch(brow[0] + kt + 32, 0, 1);
#pragma unroll
        for (int ni = 0; ni < 4; ++ni)
            c[ni] = wmma_bf(a, bfr[ni], c[ni]);
    }
#pragma unroll
    for (int ni = 0; ni < 4; ++ni) {
        const int fcol = tn * 64 + ni * 16 + nl;
        const float bias = b1[(size_t)e * DFf + fcol];
#pragma unroll
        for (int r = 0; r < 8; ++r) {
            int ml = 8 * hh + r;
            if (slot * 16 + ml < cnt) {
                int tk = list[e * Nn + slot * 16 + ml];
                float vv = c[ni][r] + bias;
                if (vv < 0.f) vv = 0.f;
                hid[(size_t)tk * DFf + fcol] = f2bf(vv);
            }
        }
    }
}

// ---------------- MoE GEMM2 + residual: 16x64 tile/wave ----------------

__global__ __launch_bounds__(32)
void moe_ffn2(const u16* __restrict__ hid, const u16* __restrict__ w2T,
              const float* __restrict__ b2, const int* __restrict__ counts,
              const int* __restrict__ list, const float* __restrict__ x1,
              float* __restrict__ out) {
    const int tn = blockIdx.x;                 // 0..11 (D/64)
    const int e = blockIdx.y >> 7;
    const int slot = blockIdx.y & 127;
    const int cnt = counts[e];
    if (slot * 16 >= cnt) return;
    const int lane = threadIdx.x & 31, nl = lane & 15, hh = lane >> 4;
    const int ridx = slot * 16 + nl;
    const int tokA = list[e * Nn + ((ridx < cnt) ? ridx : slot * 16)];
    const u16* arow = hid + (size_t)tokA * DFf;
    const u16* brow[4];
#pragma unroll
    for (int i = 0; i < 4; ++i)
        brow[i] = w2T + ((size_t)e * Dd + tn * 64 + i * 16 + nl) * DFf;
    v8f c[4] = {};
    for (int kt = 0; kt < DFf; kt += 32) {
        Frag a, bfr[4];
        load_frag(arow + kt, a);
#pragma unroll
        for (int i = 0; i < 4; ++i) load_frag(brow[i] + kt, bfr[i]);
        __builtin_prefetch(brow[0] + kt + 32, 0, 1);
#pragma unroll
        for (int ni = 0; ni < 4; ++ni)
            c[ni] = wmma_bf(a, bfr[ni], c[ni]);
    }
#pragma unroll
    for (int ni = 0; ni < 4; ++ni) {
        const int col = tn * 64 + ni * 16 + nl;
        const float bias = b2[(size_t)e * Dd + col];
#pragma unroll
        for (int r = 0; r < 8; ++r) {
            int ml = 8 * hh + r;
            if (slot * 16 + ml < cnt) {
                int tk = list[e * Nn + slot * 16 + ml];
                out[(size_t)tk * Dd + col] = c[ni][r] + bias + x1[(size_t)tk * Dd + col];
            }
        }
    }
}

// ---------------- launch ----------------

extern "C" void kernel_launch(void* const* d_in, const int* in_sizes, int n_in,
                              void* d_out, int out_size, void* d_ws, size_t ws_size,
                              hipStream_t stream) {
    const float* x     = (const float*)d_in[0];
    const float* ln1_g = (const float*)d_in[1];
    const float* ln1_b = (const float*)d_in[2];
    const float* ln2_g = (const float*)d_in[3];
    const float* ln2_b = (const float*)d_in[4];
    const float* Wq    = (const float*)d_in[5];
    const float* Wk    = (const float*)d_in[6];
    const float* Wv    = (const float*)d_in[7];
    const float* Wp    = (const float*)d_in[8];
    const float* bp    = (const float*)d_in[9];
    const float* Wg    = (const float*)d_in[10];
    const float* W1    = (const float*)d_in[11];
    const float* b1    = (const float*)d_in[12];
    const float* W2    = (const float*)d_in[13];
    const float* b2    = (const float*)d_in[14];
    float* out = (float*)d_out;

    char* ws = (char*)d_ws;
    size_t off = 0;
    auto take = [&](size_t bytes) { size_t o = off; off += (bytes + 255) & ~(size_t)255; return o; };
    u16* h1     = (u16*)(ws + take((size_t)Nn * Dd * 2));
    u16* wqkvT  = (u16*)(ws + take((size_t)NCc * Dd * 2));
    u16* wpT    = (u16*)(ws + take((size_t)Dd * Dd * 2));
    u16* w1T    = (u16*)(ws + take((size_t)Ee * DFf * Dd * 2));
    u16* w2T    = (u16*)(ws + take((size_t)Ee * Dd * DFf * 2));
    u16* qb     = (u16*)(ws + take((size_t)Nn * Dd * 2));
    u16* kb     = (u16*)(ws + take((size_t)Nn * Dd * 2));
    u16* vTb    = (u16*)(ws + take((size_t)Nn * Dd * 2));
    u16* attout = (u16*)(ws + take((size_t)Nn * Dd * 2));
    float* x1   = (float*)(ws + take((size_t)Nn * Dd * 4));
    u16* h2     = (u16*)(ws + take((size_t)Nn * Dd * 2));
    u16* hid    = (u16*)(ws + take((size_t)Nn * DFf * 2));
    int* counts = (int*)(ws + take((size_t)Ee * 4));
    int* list   = (int*)(ws + take((size_t)Ee * Nn * 4));
    (void)ws_size; (void)in_sizes; (void)n_in; (void)out_size;

    // weight prep (bf16, transposed)
    {
        long long tot = (long long)NCc * Dd;
        wt_qkv_kernel<<<(unsigned)((tot + 255) / 256), 256, 0, stream>>>(Wq, Wk, Wv, wqkvT);
    }
    wt_transpose_kernel<<<dim3((Dd * Dd + 255) / 256, 1), 256, 0, stream>>>(Wp, wpT, Dd, Dd);
    wt_transpose_kernel<<<dim3((Dd * DFf + 255) / 256, Ee), 256, 0, stream>>>(W1, w1T, Dd, DFf);
    wt_transpose_kernel<<<dim3((DFf * Dd + 255) / 256, Ee), 256, 0, stream>>>(W2, w2T, DFf, Dd);

    // attention path
    ln_kernel<<<Nn, 256, 0, stream>>>(x, ln1_g, ln1_b, h1);
    gemm_qkv<<<dim3(NCc / 64, Nn / 64), 32, 0, stream>>>(h1, wqkvT, qb, kb, vTb);
    attn_kernel<<<dim3(Tt / 16, Bb * Hh), 32, 0, stream>>>(qb, kb, vTb, attout);
    gemm_proj<<<dim3(Dd / 64, Nn / 64), 32, 0, stream>>>(attout, wpT, x, bp, x1);

    // MoE path
    ln_kernel<<<Nn, 256, 0, stream>>>(x1, ln2_g, ln2_b, h2);
    hipMemsetAsync(counts, 0, Ee * sizeof(int), stream);
    gate_kernel<<<Nn / 256, 256, 0, stream>>>(h2, Wg, counts, list);
    moe_ffn1<<<dim3(DFf / 64, Ee * (Nn / 16)), 32, 0, stream>>>(h2, w1T, b1, counts, list, hid);
    moe_ffn2<<<dim3(Dd / 64, Ee * (Nn / 16)), 32, 0, stream>>>(hid, w2T, b2, counts, list, x1, out);
}
